// Transcoders_35450660061627
// MI455X (gfx1250) — matrix operand order repaired and checked by reading, not compiled, not run on hardware
//
#include <hip/hip_runtime.h>
#include <cstdint>
#include <cstddef>

// ---------------------------------------------------------------------------
// Problem dims (fixed by the reference)
// ---------------------------------------------------------------------------
#define B_DIM 4096
#define D_DIM 768
#define L_DIM 12288

// GEMM tiling
#define BM 128
#define BN 128
#define BK 32
#define LDSTR 40   // padded LDS row stride in ushorts (32 + 8): conflict-free b128 reads

// ---------------------------------------------------------------------------
// CDNA5 WMMA types (wave32; 16x16x32 bf16 -> f32)
// ---------------------------------------------------------------------------
typedef __attribute__((ext_vector_type(16))) __bf16 v16bf;
typedef __attribute__((ext_vector_type(8)))  float  v8f;

struct __align__(16) U4 { unsigned x, y, z, w; };
struct __align__(16) Pack32 { U4 lo, hi; };

__device__ __forceinline__ unsigned short f2bf(float f) {
  unsigned u = __builtin_bit_cast(unsigned, f);
  u = u + 0x7FFFu + ((u >> 16) & 1u);          // round-to-nearest-even
  return (unsigned short)(u >> 16);
}

// ---------------------------------------------------------------------------
// f32 -> bf16 conversion (element-wise)
// ---------------------------------------------------------------------------
__global__ __launch_bounds__(256) void cvt_bf16_kernel(const float* __restrict__ in,
                                                       unsigned short* __restrict__ out,
                                                       int n) {
  int i = blockIdx.x * 256 + threadIdx.x;
  int stride = gridDim.x * 256;
  for (; i < n; i += stride) out[i] = f2bf(in[i]);
}

// W_skip [D,D] row-major -> sbT [N=D, K=D] bf16 (transpose so K is contiguous)
__global__ void transpose_cvt_kernel(const float* __restrict__ in,
                                     unsigned short* __restrict__ out) {
  int k = blockIdx.x * 16 + threadIdx.x;
  int n = blockIdx.y * 16 + threadIdx.y;
  out[(size_t)n * D_DIM + k] = f2bf(in[(size_t)k * D_DIM + n]);
}

// ---------------------------------------------------------------------------
// Zero mean_encoded region of d_out + l0 accumulator
// ---------------------------------------------------------------------------
__global__ __launch_bounds__(256) void zero_init_kernel(float* __restrict__ me,
                                                        unsigned long long n,
                                                        unsigned* __restrict__ l0_acc) {
  unsigned long long i = (unsigned long long)blockIdx.x * 256ull + threadIdx.x;
  unsigned long long stride = (unsigned long long)gridDim.x * 256ull;
  float4* p = (float4*)me;
  unsigned long long n4 = n >> 2;
  for (; i < n4; i += stride) p[i] = make_float4(0.f, 0.f, 0.f, 0.f);
  if (blockIdx.x == 0 && threadIdx.x == 0) *l0_acc = 0u;
}

// ---------------------------------------------------------------------------
// LDS-tiled WMMA GEMM:  out[M,N] = A[M,K](bf16) * Bt[N,K](bf16)^T + bias[N]
//
// Block (8 waves / 256 threads) computes a 128x128 tile. Per 32-wide K chunk:
//   - A(128x32) and B(128x32) bf16 staged in LDS (padded stride, b128 ops)
//   - next chunk prefetched into registers while WMMAs consume current stage
//   - each wave owns a 64(M) x 32(N) sub-tile: 4 A-frags x 2 B-frags -> 8 WMMA
// Global reuse factor 128 vs the naive per-wave streaming (9.4 GB -> 1.2 GB
// for the encoder GEMM at 23.3 TB/s HBM).
// Fragment lane mapping per CDNA5 ISA 7.12.2.
// ---------------------------------------------------------------------------
__global__ __launch_bounds__(256) void wmma_gemm_bt_kernel(
    const unsigned short* __restrict__ A,    // [M,K] bf16 bits
    const unsigned short* __restrict__ Bt,   // [N,K] bf16 bits
    const float* __restrict__ bias,          // [N]
    float* __restrict__ out,                 // [M,N] f32
    int N, int K) {
  __shared__ __align__(16) unsigned short As[BM * LDSTR];
  __shared__ __align__(16) unsigned short Bs[BN * LDSTR];

  const int t    = threadIdx.x;
  const int lane = t & 31;
  const int wave = t >> 5;
  const int half = lane >> 4;      // 0: lanes 0-15, 1: lanes 16-31
  const int l16  = lane & 15;

  const int m0 = blockIdx.y * BM;
  const int n0 = blockIdx.x * BN;
  const int m_blk = (wave & 1) * 64;   // wave's M offset inside block tile
  const int n_blk = (wave >> 1) * 32;  // wave's N offset inside block tile

  // cooperative stage-load mapping: thread covers rows ra and ra+64, 8-elem chunk ca
  const int ra = t >> 2;
  const int ca = (t & 3) * 8;
  const unsigned short* Ag0 = A  + (size_t)(m0 + ra)      * K + ca;
  const unsigned short* Ag1 = A  + (size_t)(m0 + ra + 64) * K + ca;
  const unsigned short* Bg0 = Bt + (size_t)(n0 + ra)      * K + ca;
  const unsigned short* Bg1 = Bt + (size_t)(n0 + ra + 64) * K + ca;

  // prefetch chunk 0
  U4 pa0 = *(const U4*)(Ag0);
  U4 pa1 = *(const U4*)(Ag1);
  U4 pb0 = *(const U4*)(Bg0);
  U4 pb1 = *(const U4*)(Bg1);

  v8f acc[4][2];
#pragma unroll
  for (int ms = 0; ms < 4; ++ms)
#pragma unroll
    for (int ns = 0; ns < 2; ++ns) acc[ms][ns] = (v8f){};

  for (int k0 = 0; k0 < K; k0 += BK) {
    // commit prefetched chunk to LDS
    *(U4*)(&As[ra * LDSTR + ca])        = pa0;
    *(U4*)(&As[(ra + 64) * LDSTR + ca]) = pa1;
    *(U4*)(&Bs[ra * LDSTR + ca])        = pb0;
    *(U4*)(&Bs[(ra + 64) * LDSTR + ca]) = pb1;
    __syncthreads();

    // issue next-chunk global loads; they fly while WMMAs run below
    const int kn = k0 + BK;
    if (kn < K) {
      pa0 = *(const U4*)(Ag0 + kn);
      pa1 = *(const U4*)(Ag1 + kn);
      pb0 = *(const U4*)(Bg0 + kn);
      pb1 = *(const U4*)(Bg1 + kn);
    }

    // load fragments from LDS
    // A: lanes<16 hold K = {0..7,16..23}; lanes>=16 hold K = {8..15,24..31}
    v16bf af[4];
#pragma unroll
    for (int ms = 0; ms < 4; ++ms) {
      const unsigned short* ap = &As[(m_blk + ms * 16 + l16) * LDSTR + half * 8];
      af[ms] = __builtin_bit_cast(v16bf, Pack32{*(const U4*)(ap), *(const U4*)(ap + 16)});
    }
    // B: lane holds column n_blk+ns*16+l16, K = half*8 + {0..7,16..23}
    v16bf bfr[2];
#pragma unroll
    for (int ns = 0; ns < 2; ++ns) {
      const unsigned short* bp = &Bs[(n_blk + ns * 16 + l16) * LDSTR + half * 8];
      bfr[ns] = __builtin_bit_cast(v16bf, Pack32{*(const U4*)(bp), *(const U4*)(bp + 16)});
    }

#pragma unroll
    for (int ms = 0; ms < 4; ++ms)
#pragma unroll
      for (int ns = 0; ns < 2; ++ns)
        acc[ms][ns] = __builtin_amdgcn_wmma_f32_16x16x32_bf16(
            false, af[ms], false, bfr[ns], (short)0, acc[ms][ns], false, false);

    __syncthreads();
  }

  // epilogue: C/D layout VGPR r -> row r (lanes 0-15) / r+8 (lanes 16-31)
#pragma unroll
  for (int ns = 0; ns < 2; ++ns) {
    const int n = n0 + n_blk + ns * 16 + l16;
    const float bn = bias[n];
#pragma unroll
    for (int ms = 0; ms < 4; ++ms) {
      const int mb = m0 + m_blk + ms * 16 + half * 8;
#pragma unroll
      for (int r = 0; r < 8; ++r) {
        out[(size_t)(mb + r) * N + n] = acc[ms][ns][r] + bn;
      }
    }
  }
}

// ---------------------------------------------------------------------------
// Per-row exact top-k via 32-bit radix select (4x 8-bit rounds) in LDS.
// One 256-thread block per row. Produces the compacted top-128 list with an
// "in top-32" flag in the index sign bit, scatters mean_encoded, counts L0.
// ---------------------------------------------------------------------------
__global__ __launch_bounds__(256) void topk_kernel(const float* __restrict__ pre,
                                                   float* __restrict__ me_out,
                                                   int* __restrict__ tk_idx,
                                                   float* __restrict__ tk_val,
                                                   unsigned* __restrict__ l0_acc) {
  __shared__ unsigned keys[L_DIM];      // 48 KB
  __shared__ unsigned hist[256];
  __shared__ unsigned s_digit, s_need;
  __shared__ unsigned s_tie128, s_tie32, s_ptr;
  __shared__ int      list_idx[128];
  __shared__ float    list_val[128];
  __shared__ unsigned redu[256];

  const int row = blockIdx.x;
  const float* prow = pre + (size_t)row * L_DIM;

  // monotonic key transform: larger float <-> larger unsigned
  for (int i = threadIdx.x; i < L_DIM; i += 256) {
    unsigned u = __builtin_bit_cast(unsigned, prow[i]);
    keys[i] = (u & 0x80000000u) ? ~u : (u | 0x80000000u);
  }
  if (threadIdx.x == 0) { s_tie128 = 0; s_tie32 = 0; s_ptr = 0; }
  __syncthreads();

  unsigned T[2], needR[2];
  const unsigned Ks[2] = {128u, 32u};
  for (int pass = 0; pass < 2; ++pass) {
    unsigned prefix = 0, mask = 0, need = Ks[pass];
    for (int shift = 24; shift >= 0; shift -= 8) {
      hist[threadIdx.x] = 0;
      __syncthreads();
      for (int i = threadIdx.x; i < L_DIM; i += 256) {
        unsigned u = keys[i];
        if ((u & mask) == prefix) atomicAdd(&hist[(u >> shift) & 255u], 1u);
      }
      __syncthreads();
      if (threadIdx.x == 0) {
        unsigned cum = 0, d = 255;
        for (;;) {
          unsigned h = hist[d];
          if (cum + h >= need || d == 0) { s_digit = d; s_need = need - cum; break; }
          cum += h; --d;
        }
      }
      __syncthreads();
      prefix |= (s_digit << shift);
      mask   |= (0xFFu << shift);
      need    = s_need;
      __syncthreads();
    }
    T[pass] = prefix;      // exact threshold key
    needR[pass] = need;    // how many ties at T to take
  }

  const unsigned T128 = T[0], need128 = needR[0];
  const unsigned T32  = T[1], need32  = needR[1];

  unsigned local_l0 = 0;
  for (int i = threadIdx.x; i < L_DIM; i += 256) {
    unsigned u = keys[i];
    bool sel128 = false;
    if (u > T128)        sel128 = true;
    else if (u == T128)  sel128 = (atomicAdd(&s_tie128, 1u) < need128);
    if (sel128) {
      bool sel32 = false;
      if (u > T32)       sel32 = true;
      else if (u == T32) sel32 = (atomicAdd(&s_tie32, 1u) < need32);
      unsigned p = atomicAdd(&s_ptr, 1u);
      unsigned fb = (u & 0x80000000u) ? (u ^ 0x80000000u) : ~u;  // inverse key
      float f = __builtin_bit_cast(float, fb);
      float rv = f > 0.0f ? f : 0.0f;                             // ReLU
      list_idx[p] = (int)((unsigned)i | (sel32 ? 0x80000000u : 0u));
      list_val[p] = rv;
      if (rv > 0.0f) {
        ++local_l0;
        // mean over k in {32,128}: top32 entries count twice
        me_out[(size_t)row * L_DIM + i] = rv * (sel32 ? 1.0f : 0.5f);
      }
    }
  }
  __syncthreads();

  if (threadIdx.x < 128) {
    tk_idx[(size_t)row * 128 + threadIdx.x] = list_idx[threadIdx.x];
    tk_val[(size_t)row * 128 + threadIdx.x] = list_val[threadIdx.x];
  }
  redu[threadIdx.x] = local_l0;
  __syncthreads();
  for (int s = 128; s > 0; s >>= 1) {
    if (threadIdx.x < s) redu[threadIdx.x] += redu[threadIdx.x + s];
    __syncthreads();
  }
  if (threadIdx.x == 0) atomicAdd(l0_acc, redu[0]);
}

// ---------------------------------------------------------------------------
// Sparse decode for both k values, fused with mean_decoded + loss partial.
// One 256-thread block per batch row; each thread owns 3 of 768 columns.
// ---------------------------------------------------------------------------
__global__ __launch_bounds__(256) void decode_kernel(const int* __restrict__ tk_idx,
                                                     const float* __restrict__ tk_val,
                                                     const float* __restrict__ skip,
                                                     const float* __restrict__ decoder,
                                                     const float* __restrict__ mlp,
                                                     float* __restrict__ md_out,
                                                     float* __restrict__ loss_part) {
  __shared__ int   s_idx[128];
  __shared__ float s_val[128];
  __shared__ float red[256];
  const int row = blockIdx.x;
  if (threadIdx.x < 128) {
    s_idx[threadIdx.x] = tk_idx[(size_t)row * 128 + threadIdx.x];
    s_val[threadIdx.x] = tk_val[(size_t)row * 128 + threadIdx.x];
  }
  __syncthreads();

  float a32[3], a128[3];
#pragma unroll
  for (int j = 0; j < 3; ++j) {
    float s = skip[(size_t)row * D_DIM + threadIdx.x + 256 * j];
    a32[j] = s; a128[j] = s;
  }
  for (int t = 0; t < 128; ++t) {
    float v = s_val[t];
    if (v == 0.0f) continue;               // ReLU'd-away feature contributes nothing
    int packed = s_idx[t];
    int idx = packed & 0x7FFFFFFF;
    bool in32 = packed < 0;
    const float* drow = decoder + (size_t)idx * D_DIM;
#pragma unroll
    for (int j = 0; j < 3; ++j) {
      float d = drow[threadIdx.x + 256 * j];
      a128[j] += v * d;
      if (in32) a32[j] += v * d;
    }
  }

  float lsum = 0.f;
#pragma unroll
  for (int j = 0; j < 3; ++j) {
    int c = threadIdx.x + 256 * j;
    float m = mlp[(size_t)row * D_DIM + c];
    md_out[(size_t)row * D_DIM + c] = 0.5f * (a32[j] + a128[j]);
    float e32 = m - a32[j], e128 = m - a128[j];
    lsum += e32 * e32 + e128 * e128;
  }
  red[threadIdx.x] = lsum;
  __syncthreads();
  for (int s = 128; s > 0; s >>= 1) {
    if (threadIdx.x < s) red[threadIdx.x] += red[threadIdx.x + s];
    __syncthreads();
  }
  if (threadIdx.x == 0) loss_part[row] = red[0];
}

// ---------------------------------------------------------------------------
// total_variance partial per column: sum_b (x[b,c] - mean_c)^2
// ---------------------------------------------------------------------------
__global__ __launch_bounds__(256) void variance_kernel(const float* __restrict__ x,
                                                       float* __restrict__ var_part) {
  __shared__ float red[256];
  const int c = blockIdx.x;
  float s = 0.f;
  for (int b = threadIdx.x; b < B_DIM; b += 256) s += x[(size_t)b * D_DIM + c];
  red[threadIdx.x] = s;
  __syncthreads();
  for (int k = 128; k > 0; k >>= 1) {
    if (threadIdx.x < k) red[threadIdx.x] += red[threadIdx.x + k];
    __syncthreads();
  }
  const float mean = red[0] / (float)B_DIM;
  __syncthreads();
  float v = 0.f;
  for (int b = threadIdx.x; b < B_DIM; b += 256) {
    float d = x[(size_t)b * D_DIM + c] - mean;
    v += d * d;
  }
  red[threadIdx.x] = v;
  __syncthreads();
  for (int k = 128; k > 0; k >>= 1) {
    if (threadIdx.x < k) red[threadIdx.x] += red[threadIdx.x + k];
    __syncthreads();
  }
  if (threadIdx.x == 0) var_part[c] = red[0];
}

// ---------------------------------------------------------------------------
// Deterministic final reductions: loss / tv and l0 count
// ---------------------------------------------------------------------------
__global__ __launch_bounds__(256) void finalize_kernel(const float* __restrict__ loss_part,
                                                       const float* __restrict__ var_part,
                                                       const unsigned* __restrict__ l0_acc,
                                                       float* __restrict__ out2) {
  __shared__ float red[256];
  float s = 0.f;
  for (int i = threadIdx.x; i < B_DIM; i += 256) s += loss_part[i];
  red[threadIdx.x] = s;
  __syncthreads();
  for (int k = 128; k > 0; k >>= 1) {
    if (threadIdx.x < k) red[threadIdx.x] += red[threadIdx.x + k];
    __syncthreads();
  }
  float loss = red[0];
  __syncthreads();
  float t = 0.f;
  for (int i = threadIdx.x; i < D_DIM; i += 256) t += var_part[i];
  red[threadIdx.x] = t;
  __syncthreads();
  for (int k = 128; k > 0; k >>= 1) {
    if (threadIdx.x < k) red[threadIdx.x] += red[threadIdx.x + k];
    __syncthreads();
  }
  if (threadIdx.x == 0) {
    out2[0] = loss / red[0];
    out2[1] = (float)(*l0_acc);
  }
}

// ---------------------------------------------------------------------------
// Host-side orchestration
// ---------------------------------------------------------------------------
extern "C" void kernel_launch(void* const* d_in, const int* in_sizes, int n_in,
                              void* d_out, int out_size, void* d_ws, size_t ws_size,
                              hipStream_t stream) {
  (void)in_sizes; (void)n_in; (void)out_size; (void)ws_size;

  const float* x       = (const float*)d_in[0];   // [B,D]
  const float* mlp     = (const float*)d_in[1];   // [B,D]
  const float* enc_W   = (const float*)d_in[2];   // [L,D]
  const float* enc_b   = (const float*)d_in[3];   // [L]
  const float* decoder = (const float*)d_in[4];   // [L,D]
  const float* b_dec   = (const float*)d_in[5];   // [D]
  const float* W_skip  = (const float*)d_in[6];   // [D,D]

  float* out_me = (float*)d_out;                        // [B,L] mean_encoded
  float* out_md = out_me + (size_t)B_DIM * L_DIM;       // [B,D] mean_decoded
  float* out_sc = out_md + (size_t)B_DIM * D_DIM;       // [loss, l0]

  // workspace carve-up (256B aligned slots)
  char* w = (char*)d_ws;
  auto carve = [&](size_t bytes) {
    char* p = w;
    w += (bytes + 255) & ~(size_t)255;
    return p;
  };
  unsigned short* xb   = (unsigned short*)carve((size_t)B_DIM * D_DIM * 2);
  unsigned short* wb   = (unsigned short*)carve((size_t)L_DIM * D_DIM * 2);
  unsigned short* sbT  = (unsigned short*)carve((size_t)D_DIM * D_DIM * 2);
  float* pre           = (float*)carve((size_t)B_DIM * L_DIM * 4);
  float* skip          = (float*)carve((size_t)B_DIM * D_DIM * 4);
  int*   tk_idx        = (int*)carve((size_t)B_DIM * 128 * 4);
  float* tk_val        = (float*)carve((size_t)B_DIM * 128 * 4);
  float* loss_part     = (float*)carve((size_t)B_DIM * 4);
  float* var_part      = (float*)carve((size_t)D_DIM * 4);
  unsigned* l0_acc     = (unsigned*)carve(256);

  // 1) precision prep: bf16 copies of x, enc_W; transposed bf16 W_skip
  int n1 = B_DIM * D_DIM;
  cvt_bf16_kernel<<<(n1 + 255) / 256, 256, 0, stream>>>(x, xb, n1);
  int n2 = L_DIM * D_DIM;
  cvt_bf16_kernel<<<(n2 + 255) / 256, 256, 0, stream>>>(enc_W, wb, n2);
  transpose_cvt_kernel<<<dim3(D_DIM / 16, D_DIM / 16), dim3(16, 16), 0, stream>>>(W_skip, sbT);

  // 2) zero sparse mean_encoded output + l0 accumulator
  zero_init_kernel<<<4096, 256, 0, stream>>>(out_me, (unsigned long long)B_DIM * L_DIM, l0_acc);

  // 3) encoder GEMM: pre = x @ enc_W^T + enc_b   (LDS-tiled WMMA bf16 -> f32)
  wmma_gemm_bt_kernel<<<dim3(L_DIM / BN, B_DIM / BM), 256, 0, stream>>>(
      xb, wb, enc_b, pre, L_DIM, D_DIM);

  // 4) skip GEMM: skip = x @ W_skip + b_dec      (LDS-tiled WMMA bf16 -> f32)
  wmma_gemm_bt_kernel<<<dim3(D_DIM / BN, B_DIM / BM), 256, 0, stream>>>(
      xb, sbT, b_dec, skip, D_DIM, D_DIM);

  // 5) total_variance partials
  variance_kernel<<<D_DIM, 256, 0, stream>>>(x, var_part);

  // 6) per-row exact top-128/top-32 select, mean_encoded scatter, l0 count
  topk_kernel<<<B_DIM, 256, 0, stream>>>(pre, out_me, tk_idx, tk_val, l0_acc);

  // 7) sparse decode (both k), mean_decoded, loss partials
  decode_kernel<<<B_DIM, 256, 0, stream>>>(tk_idx, tk_val, skip, decoder, mlp,
                                           out_md, loss_part);

  // 8) deterministic scalar finalize
  finalize_kernel<<<1, 256, 0, stream>>>(loss_part, var_part, l0_acc, out_sc);
}
